// AttLayer_89258010345811
// MI455X (gfx1250) — matrix operand order repaired and checked by reading, not compiled, not run on hardware
//
#include <hip/hip_runtime.h>
#include <math.h>

typedef __attribute__((ext_vector_type(16))) _Float16 v16h;
typedef __attribute__((ext_vector_type(8)))  _Float16 v8h;
typedef __attribute__((ext_vector_type(8)))  float    v8f;

#define C_IN   512
#define C_RED  256
#define LSEQ   16384
#define BLK    64
#define NBLK   256      // LSEQ / BLK

// LDS strides (elements, f16). All are multiples of 8 (16B alignment) and
// chosen so row-stride in dwords mod 64 == 4 or 36 -> conflict-free b128 lane loads.
#define XT_RS  520      // X^T  [64 cols][512 ch]
#define QT_RS  264      // Q^T  [64 cols][256 ch]   (also K^T, H^T)
#define VT_RS  72       // V    [256 ch][64 cols]   (also P [64][64])

#define XT_OFF 0
#define PT_OFF 0                         // P reuses X region after phase 2
#define QT_OFF (64 * XT_RS * 2)          // 66560
#define HT_OFF QT_OFF                    // H^T reuses Q region after scores
#define KT_OFF (QT_OFF + 64 * QT_RS * 2) // 100352
#define VT_OFF (KT_OFF + 64 * QT_RS * 2) // 134144
#define LDS_BYTES (VT_OFF + 256 * VT_RS * 2) // 171008

static __device__ __forceinline__ v16h cat8(v8h lo, v8h hi) {
  return __builtin_shufflevector(lo, hi, 0,1,2,3,4,5,6,7,8,9,10,11,12,13,14,15);
}

// A fragment (16x32 f16, M x K): lane m = l&15, half = l>>4.
// lanes 0-15 hold K[0..7]+K[16..23], lanes 16-31 hold K[8..15]+K[24..31] (ISA 7.12.2)
static __device__ __forceinline__ v16h load_a16(const _Float16* t, int rs, int lane) {
  int m = lane & 15, h = lane >> 4;
  const _Float16* p = t + m * rs + 8 * h;
  v8h lo = *(const v8h*)(p);
  v8h hi = *(const v8h*)(p + 16);
  return cat8(lo, hi);
}

// B fragment (32x16 f16, K x N) from [N][K] storage: lane n = l&15, half = l>>4,
// lanes 0-15 hold K[0..7]+K[16..23], lanes 16-31 hold K[8..15]+K[24..31] (matches A)
static __device__ __forceinline__ v16h load_b16(const _Float16* t, int rs, int lane) {
  int n = lane & 15, h = lane >> 4;
  const _Float16* p = t + n * rs + 8 * h;
  v8h lo = *(const v8h*)(p);
  v8h hi = *(const v8h*)(p + 16);
  return cat8(lo, hi);
}

static __device__ __forceinline__ v8f wmma32(v16h a, v16h b, v8f c) {
  return __builtin_amdgcn_wmma_f32_16x16x32_f16(false, a, false, b, (short)0, c, false, false);
}

// D tile (M x N f32): lane holds N = l&15, rows M = r + 8*(l>>4), r = 0..7.
// Transposed store -> dst[n][m], 8 contiguous f16 = one 16B LDS store per lane.
static __device__ __forceinline__ void store_dT(_Float16* t, int rs, v8f c, int lane) {
  int n = lane & 15, h = lane >> 4;
  v8h v;
#pragma unroll
  for (int r = 0; r < 8; ++r) v[r] = (_Float16)c[r];
  *(v8h*)(t + n * rs + 8 * h) = v;
}

// Row-major store dst[m][n] (scattered f16 stores).
static __device__ __forceinline__ void store_dR(_Float16* t, int rs, v8f c, int lane,
                                                bool relu) {
  int n = lane & 15, h = lane >> 4;
#pragma unroll
  for (int r = 0; r < 8; ++r) {
    float v = c[r];
    if (relu) v = fmaxf(v, 0.0f);
    t[(8 * h + r) * rs + n] = (_Float16)v;
  }
}

__global__ void convert_weights(const float* __restrict__ Wq,
                                const float* __restrict__ Wk,
                                const float* __restrict__ Wv,
                                const float* __restrict__ Wo,
                                _Float16* __restrict__ dst) {
  int i = blockIdx.x * 256 + threadIdx.x;           // 0 .. 524287
  int off = i & 131071;
  const float* src = (i < 131072) ? Wq : (i < 262144) ? Wk : (i < 393216) ? Wv : Wo;
  dst[i] = (_Float16)src[off];
}

__global__ __launch_bounds__(256, 1)
void attn_fused(const float* __restrict__ x1,
                const float* __restrict__ maskp,
                const _Float16* __restrict__ wsh,   // WqH|WkH|WvH|WoH, 131072 each
                const float* __restrict__ bq, const float* __restrict__ bk,
                const float* __restrict__ bv, const float* __restrict__ bo,
                float* __restrict__ out) {
  extern __shared__ __align__(16) char smem[];
  _Float16* Xt = (_Float16*)(smem + XT_OFF);
  _Float16* Pt = (_Float16*)(smem + PT_OFF);
  _Float16* Qt = (_Float16*)(smem + QT_OFF);
  _Float16* Ht = (_Float16*)(smem + HT_OFF);
  _Float16* Kt = (_Float16*)(smem + KT_OFF);
  _Float16* Vt = (_Float16*)(smem + VT_OFF);

  const int tid  = threadIdx.x;
  const int lane = tid & 31;
  const int wv   = tid >> 5;            // wave id 0..7
  const int bb   = blockIdx.x / NBLK;   // batch
  const int blk  = blockIdx.x % NBLK;
  const int l0   = blk * BLK;

  // ---- Phase 1: stage X tile (512 ch x 64 cols f32) into LDS as f16 [col][ch]
  {
    const float* xsrc = x1 + ((size_t)bb * C_IN) * LSEQ + l0;
#pragma unroll 4
    for (int it = 0; it < 32; ++it) {
      int idx = it * 256 + tid;          // 0..8191 = 512 rows * 16 float4
      int c = idx >> 4;
      int q = idx & 15;
      float4 v = *(const float4*)(xsrc + (size_t)c * LSEQ + 4 * q);
      Xt[(4 * q + 0) * XT_RS + c] = (_Float16)v.x;
      Xt[(4 * q + 1) * XT_RS + c] = (_Float16)v.y;
      Xt[(4 * q + 2) * XT_RS + c] = (_Float16)v.z;
      Xt[(4 * q + 3) * XT_RS + c] = (_Float16)v.w;
    }
  }
  __syncthreads();

  // ---- Phase 2: Q/K/V = W(256x512) @ X(512x64) + b.
  // 48 row-groups (s, mt), each = 4 N-tiles sharing the same A (weight) fragments:
  // A loaded once per kk, reused across 4 WMMAs -> 4x less L2 weight traffic.
  for (int g = wv; g < 48; g += 8) {
    int s = g / 16, mt = g % 16;
    const _Float16* W = wsh + s * 131072 + (mt * 16) * C_IN;
    v8f acc[4] = {};
#pragma unroll
    for (int kk = 0; kk < 16; ++kk) {
      v16h a = load_a16(W + kk * 32, C_IN, lane);
#pragma unroll
      for (int nt = 0; nt < 4; ++nt) {
        v16h b = load_b16(Xt + (nt * 16) * XT_RS + kk * 32, XT_RS, lane);
        acc[nt] = wmma32(a, b, acc[nt]);
      }
    }
    const float* bias = (s == 0) ? bq : (s == 1) ? bk : bv;
    int mb = mt * 16 + 8 * (lane >> 4);
    float br[8];
#pragma unroll
    for (int r = 0; r < 8; ++r) br[r] = bias[mb + r];
#pragma unroll
    for (int nt = 0; nt < 4; ++nt) {
#pragma unroll
      for (int r = 0; r < 8; ++r) acc[nt][r] += br[r];
      if (s == 0)      store_dT(Qt + (nt * 16) * QT_RS + mt * 16, QT_RS, acc[nt], lane);
      else if (s == 1) store_dT(Kt + (nt * 16) * QT_RS + mt * 16, QT_RS, acc[nt], lane);
      else             store_dR(Vt + (mt * 16) * VT_RS + nt * 16, VT_RS, acc[nt], lane, false);
    }
  }
  __syncthreads();

  // ---- Phase 3: scores (64x64) = Q^T K * scale, mask, softmax -> P (waves 0-3)
  if (wv < 4) {
    int ib = wv * 16;                    // this wave's 16 query rows
    v8f sc[4] = {};
#pragma unroll
    for (int kk = 0; kk < 8; ++kk) {
      v16h a = load_a16(Qt + ib * QT_RS + kk * 32, QT_RS, lane);
#pragma unroll
      for (int nt = 0; nt < 4; ++nt) {
        v16h b = load_b16(Kt + (nt * 16) * QT_RS + kk * 32, QT_RS, lane);
        sc[nt] = wmma32(a, b, sc[nt]);
      }
    }
    const float NEGINF = -__builtin_inff();
    const float* mrow = maskp + (size_t)bb * LSEQ + l0;
#pragma unroll
    for (int nt = 0; nt < 4; ++nt) {
      float mv = mrow[nt * 16 + (lane & 15)];
#pragma unroll
      for (int r = 0; r < 8; ++r) {
        float v = sc[nt][r] * 0.0625f;   // 1/sqrt(256)
        sc[nt][r] = (mv == 0.0f) ? NEGINF : v;
      }
    }
    float mx[8], sm[8];
#pragma unroll
    for (int r = 0; r < 8; ++r) {
      float m = fmaxf(fmaxf(sc[0][r], sc[1][r]), fmaxf(sc[2][r], sc[3][r]));
#pragma unroll
      for (int off = 1; off < 16; off <<= 1)
        m = fmaxf(m, __shfl_xor(m, off, 32));   // stays inside 16-lane half = one row
      if (m < -1e30f) m = 0.0f;                 // fully-masked row guard
      mx[r] = m;
    }
#pragma unroll
    for (int nt = 0; nt < 4; ++nt)
#pragma unroll
      for (int r = 0; r < 8; ++r) sc[nt][r] = __expf(sc[nt][r] - mx[r]);
#pragma unroll
    for (int r = 0; r < 8; ++r) {
      float s = sc[0][r] + sc[1][r] + sc[2][r] + sc[3][r];
#pragma unroll
      for (int off = 1; off < 16; off <<= 1) s += __shfl_xor(s, off, 32);
      sm[r] = 1.0f / fmaxf(s, 1e-20f);
    }
    int h = lane >> 4, n = lane & 15;
#pragma unroll
    for (int nt = 0; nt < 4; ++nt)
#pragma unroll
      for (int r = 0; r < 8; ++r)
        Pt[(ib + 8 * h + r) * VT_RS + nt * 16 + n] = (_Float16)(sc[nt][r] * sm[r]);
  }
  __syncthreads();

  // ---- Phase 4: H^T (64x256) = P(64x64) @ V^T(64x256), ReLU.  64 tiles / 8 waves.
  for (int t = wv; t < 64; t += 8) {
    int mt = t >> 4, nt = t & 15;        // 4 i-tiles x 16 c-tiles
    v8f acc = {};
#pragma unroll
    for (int kk = 0; kk < 2; ++kk) {
      v16h a = load_a16(Pt + (mt * 16) * VT_RS + kk * 32, VT_RS, lane);
      v16h b = load_b16(Vt + (nt * 16) * VT_RS + kk * 32, VT_RS, lane);
      acc = wmma32(a, b, acc);
    }
    store_dR(Ht + (mt * 16) * QT_RS + nt * 16, QT_RS, acc, lane, true);
  }
  __syncthreads();

  // ---- Phase 5: out(512x64) = Wo(512x256) @ H + bo.
  // 32 row-groups, each = 4 N-tiles sharing the A (Wo) fragments.
  const _Float16* WoH = wsh + 3 * 131072;
  for (int g = wv; g < 32; g += 8) {
    int mt = g;
    v8f acc[4] = {};
#pragma unroll
    for (int kk = 0; kk < 8; ++kk) {
      v16h a = load_a16(WoH + (mt * 16) * C_RED + kk * 32, C_RED, lane);
#pragma unroll
      for (int nt = 0; nt < 4; ++nt) {
        v16h b = load_b16(Ht + (nt * 16) * QT_RS + kk * 32, QT_RS, lane);
        acc[nt] = wmma32(a, b, acc[nt]);
      }
    }
    int ob = mt * 16 + 8 * (lane >> 4), n = lane & 15;
    float br[8];
#pragma unroll
    for (int r = 0; r < 8; ++r) br[r] = bo[ob + r];
    float* obase = out + ((size_t)bb * C_IN) * LSEQ + (size_t)l0 + n;
#pragma unroll
    for (int nt = 0; nt < 4; ++nt) {
      float* op = obase + nt * 16;
#pragma unroll
      for (int r = 0; r < 8; ++r)
        op[(size_t)(ob + r) * LSEQ] = acc[nt][r] + br[r];
    }
  }
}

extern "C" void kernel_launch(void* const* d_in, const int* in_sizes, int n_in,
                              void* d_out, int out_size, void* d_ws, size_t ws_size,
                              hipStream_t stream) {
  const float* x1   = (const float*)d_in[0];
  // d_in[1] (x2) is unused by the reference
  const float* mask = (const float*)d_in[2];
  const float* Wq   = (const float*)d_in[3];
  const float* bq   = (const float*)d_in[4];
  const float* Wk   = (const float*)d_in[5];
  const float* bk   = (const float*)d_in[6];
  const float* Wv   = (const float*)d_in[7];
  const float* bv   = (const float*)d_in[8];
  const float* Wo   = (const float*)d_in[9];
  const float* bo   = (const float*)d_in[10];
  _Float16* wsh = (_Float16*)d_ws;            // 4 * 131072 f16 = 1 MB scratch
  float* outp = (float*)d_out;

  convert_weights<<<2048, 256, 0, stream>>>(Wq, Wk, Wv, Wo, wsh);
  attn_fused<<<4 * NBLK, 256, LDS_BYTES, stream>>>(x1, mask, wsh, bq, bk, bv, bo, outp);
}
